// RWKV_TimeMix_26886495273155
// MI455X (gfx1250) — matrix operand order repaired and checked by reading, not compiled, not run on hardware
//
#include <hip/hip_runtime.h>
#include <hip/hip_bf16.h>

// Problem constants (match reference)
#define B_ 32
#define T_ 256
#define C_ 1024
#define M_ (B_ * T_)   // 8192 rows

typedef __bf16 bf16_t;
typedef __attribute__((ext_vector_type(16))) __bf16 v16bf;
typedef __attribute__((ext_vector_type(8)))  __bf16 v8bf;
typedef __attribute__((ext_vector_type(8)))  float  v8f;

// ---------------------------------------------------------------------------
// Kernel 1: time-mix + f32 -> bf16 conversion of the three mixed activations.
// ---------------------------------------------------------------------------
__global__ __launch_bounds__(256)
void mix_kernel(const float* __restrict__ x,
                const float* __restrict__ mk,
                const float* __restrict__ mv,
                const float* __restrict__ mr,
                bf16_t* __restrict__ xk,
                bf16_t* __restrict__ xv,
                bf16_t* __restrict__ xr) {
    int tid = blockIdx.x * blockDim.x + threadIdx.x;   // [0, M_*C_)
    int c = tid & (C_ - 1);
    int m = tid >> 10;            // row index in [0, M_)
    int t = m & (T_ - 1);         // position within sequence
    float xc = x[tid];
    float xp = (t > 0) ? x[tid - C_] : 0.0f;
    float fk = mk[c], fv = mv[c], fr = mr[c];
    xk[tid] = (bf16_t)(xc * fk + xp * (1.0f - fk));
    xv[tid] = (bf16_t)(xc * fv + xp * (1.0f - fv));
    xr[tid] = (bf16_t)(xc * fr + xp * (1.0f - fr));
}

// ---------------------------------------------------------------------------
// Kernel 2: f32 -> bf16 weight conversion.
// ---------------------------------------------------------------------------
__global__ __launch_bounds__(256)
void cvt_kernel(const float* __restrict__ src, bf16_t* __restrict__ dst) {
    int tid = blockIdx.x * blockDim.x + threadIdx.x;
    dst[tid] = (bf16_t)src[tid];
}

// ---------------------------------------------------------------------------
// CDNA5 async-copy helpers (ASYNCcnt-tracked DMA global -> LDS, no VGPR
// round trip).  VGLOBAL GV mode: VDST = VGPR holding LDS byte address,
// VADDR = VGPR pair holding 64-bit global address.
// ---------------------------------------------------------------------------
__device__ __forceinline__ void async_ld_b128(unsigned lds_addr, const bf16_t* gaddr) {
    asm volatile("global_load_async_to_lds_b128 %0, %1, off"
                 :: "v"(lds_addr), "v"(gaddr) : "memory");
}
__device__ __forceinline__ void wait_asynccnt0() {
    asm volatile("s_wait_asynccnt 0x0" ::: "memory");
}
__device__ __forceinline__ unsigned lds_off(const void* p) {
    // LDS aperture: flat addr bits [31:0] carry the LDS byte address.
    return (unsigned)(uintptr_t)p;
}

// ---------------------------------------------------------------------------
// Kernel 3: bf16 WMMA GEMM:  Out[m,n] = sum_k A[m,k] * W[n,k]   (f32 out)
// N = K = 1024 compile-time.
//
// Block = 256 threads = 8 waves; block tile = 128(M) x 128(N); K-step 64.
// Double-buffered LDS tiles filled by global_load_async_to_lds_b128;
// tile t+1 streams in while the 8 waves run WMMAs against tile t.
// Wave grid 4x2: each wave owns 32(M) x 64(N) = 2x4 accumulators, and per
// 32-K sub-step performs 8 v_wmma_f32_16x16x32_bf16.
//
// Fragment <-> LDS mapping (CDNA5 16-bit WMMA register layouts):
//   A 16x32: lane l (ll=l&15, lh=l>>4): row=ll, K-chunks {lh*8, lh*8+16}
//   B 32x16: lane l: col=ll, K-chunks {lh*8, lh*8+16}
//   C/D   : VGPR r, lane l -> row = r + lh*8, col = ll
// LDS row stride 72 elements (144 B) keeps all 16B chunks aligned and
// staggers bank groups.  EXEC all-ones throughout (WMMA requirement).
// ---------------------------------------------------------------------------
#define KSTEP 64
#define LDSW  72

union AFrag { v16bf v; v8bf h[2]; };

__device__ __forceinline__ void load_tile_async(const bf16_t* __restrict__ gbase,
                                                bf16_t* lbase, int tid) {
    // 128 rows x 64 cols bf16 = 16 KB = 1024 x 16B chunks; 4 per thread.
#pragma unroll
    for (int i = 0; i < 4; ++i) {
        int id    = tid + 256 * i;
        int row   = id >> 3;        // 0..127
        int chunk = id & 7;         // 8-element (16B) chunk within 64 cols
        async_ld_b128(lds_off(lbase + row * LDSW + chunk * 8),
                      gbase + (size_t)row * C_ + chunk * 8);
    }
}

__global__ __launch_bounds__(256)
void gemm_bf16_wmma(const bf16_t* __restrict__ A,
                    const bf16_t* __restrict__ W,
                    float* __restrict__ Out) {
    __shared__ bf16_t sA[2][128 * LDSW];
    __shared__ bf16_t sB[2][128 * LDSW];

    const int tid   = threadIdx.x;
    const int lane  = tid & 31;
    const int wave  = tid >> 5;
    const int waveM = wave & 3;        // 4 waves along M
    const int waveN = wave >> 2;       // 2 waves along N
    const int lh    = lane >> 4;       // lane half (0/1)
    const int ll    = lane & 15;
    const int blockM = blockIdx.x * 128;
    const int blockN = blockIdx.y * 128;

    const bf16_t* Ab = A + (size_t)blockM * C_;   // block's A rows
    const bf16_t* Wb = W + (size_t)blockN * C_;   // block's W rows (= B cols)

    v8f acc[2][4] = {};

    // Prime buffer 0 with the first K-tile.
    load_tile_async(Ab, &sA[0][0], tid);
    load_tile_async(Wb, &sB[0][0], tid);
    wait_asynccnt0();
    __syncthreads();

    int p = 0;
    for (int k0 = 0; k0 < C_; k0 += KSTEP) {
        // Stream next tile into the other buffer while we compute.
        if (k0 + KSTEP < C_) {
            load_tile_async(Ab + k0 + KSTEP, &sA[p ^ 1][0], tid);
            load_tile_async(Wb + k0 + KSTEP, &sB[p ^ 1][0], tid);
        }

#pragma unroll
        for (int ksub = 0; ksub < KSTEP; ksub += 32) {
            AFrag af[2];
#pragma unroll
            for (int i = 0; i < 2; ++i) {
                const bf16_t* ap =
                    &sA[p][(waveM * 32 + i * 16 + ll) * LDSW + ksub + lh * 8];
                af[i].h[0] = *(const v8bf*)(ap);
                af[i].h[1] = *(const v8bf*)(ap + 16);
            }
            AFrag bfrag[4];
#pragma unroll
            for (int j = 0; j < 4; ++j) {
                const bf16_t* bp =
                    &sB[p][(waveN * 64 + j * 16 + ll) * LDSW + ksub + lh * 8];
                bfrag[j].h[0] = *(const v8bf*)(bp);
                bfrag[j].h[1] = *(const v8bf*)(bp + 16);
            }
#pragma unroll
            for (int i = 0; i < 2; ++i)
#pragma unroll
                for (int j = 0; j < 4; ++j)
                    acc[i][j] = __builtin_amdgcn_wmma_f32_16x16x32_bf16(
                        false, af[i].v, false, bfrag[j].v,
                        (short)0, acc[i][j], false, false);
        }

        // Own async loads done -> barrier makes the new tile visible to all.
        wait_asynccnt0();
        __syncthreads();
        p ^= 1;
    }

#pragma unroll
    for (int i = 0; i < 2; ++i) {
#pragma unroll
        for (int j = 0; j < 4; ++j) {
            float* op = Out + (size_t)(blockM + waveM * 32 + i * 16 + lh * 8) * C_
                            + blockN + waveN * 64 + j * 16 + ll;
#pragma unroll
            for (int r = 0; r < 8; ++r)
                op[(size_t)r * C_] = acc[i][j][r];
        }
    }
}

// ---------------------------------------------------------------------------
// Kernel 4: numerically-stabilized WKV scan + sigmoid(r) multiply.
// One lane per (batch, channel): 32768 independent recurrences over T=256.
// ---------------------------------------------------------------------------
__global__ __launch_bounds__(256)
void wkv_sig_kernel(const float* __restrict__ td,
                    const float* __restrict__ tfirst,
                    const float* __restrict__ k,
                    const float* __restrict__ v,
                    const float* __restrict__ r,
                    bf16_t* __restrict__ a) {
    int ch = blockIdx.x * blockDim.x + threadIdx.x;    // [0, B_*C_)
    int b = ch >> 10;
    int c = ch & (C_ - 1);
    float w = -__expf(td[c]);
    float u = tfirst[c];
    float p = 0.0f, q = 0.0f, o = -1e38f;
    size_t idx = (size_t)b * T_ * C_ + c;
    for (int t = 0; t < T_; ++t, idx += C_) {
        float kt = k[idx];
        float vt = v[idx];
        float no  = fmaxf(o, u + kt);
        float Aa  = __expf(o - no);
        float Bb  = __expf(u + kt - no);
        float y   = (Aa * p + Bb * vt) / (Aa * q + Bb);
        float rr  = r[idx];
        float sr  = 1.0f / (1.0f + __expf(-rr));
        a[idx] = (bf16_t)(sr * y);
        float no2 = fmaxf(w + o, kt);
        float A2  = __expf(w + o - no2);
        float B2  = __expf(kt - no2);
        p = A2 * p + B2 * vt;
        q = A2 * q + B2;
        o = no2;
    }
}

// ---------------------------------------------------------------------------
// Host-side orchestration.  Workspace ~168 MB, fixed offsets.
// ---------------------------------------------------------------------------
extern "C" void kernel_launch(void* const* d_in, const int* in_sizes, int n_in,
                              void* d_out, int out_size, void* d_ws, size_t ws_size,
                              hipStream_t stream) {
    const float* x  = (const float*)d_in[0];
    const float* td = (const float*)d_in[1];
    const float* tf = (const float*)d_in[2];
    const float* mk = (const float*)d_in[3];
    const float* mv = (const float*)d_in[4];
    const float* mr = (const float*)d_in[5];
    const float* Wk = (const float*)d_in[6];
    const float* Wv = (const float*)d_in[7];
    const float* Wr = (const float*)d_in[8];
    const float* Wo = (const float*)d_in[9];
    float* out = (float*)d_out;

    const size_t MC = (size_t)M_ * C_;      // 8,388,608
    const size_t CC = (size_t)C_ * C_;      // 1,048,576

    char* ws = (char*)d_ws;
    bf16_t* xk  = (bf16_t*)ws;                 ws += MC * 2;
    bf16_t* xv  = (bf16_t*)ws;                 ws += MC * 2;
    bf16_t* xr  = (bf16_t*)ws;                 ws += MC * 2;
    bf16_t* wkb = (bf16_t*)ws;                 ws += CC * 2;
    bf16_t* wvb = (bf16_t*)ws;                 ws += CC * 2;
    bf16_t* wrb = (bf16_t*)ws;                 ws += CC * 2;
    bf16_t* wob = (bf16_t*)ws;                 ws += CC * 2;
    float*  kf  = (float*)ws;                  ws += MC * 4;
    float*  vf  = (float*)ws;                  ws += MC * 4;
    float*  rf  = (float*)ws;                  ws += MC * 4;
    bf16_t* abf = (bf16_t*)ws;                 ws += MC * 2;

    // 1) time-mix + bf16 conversion
    mix_kernel<<<dim3((unsigned)(MC / 256)), dim3(256), 0, stream>>>(
        x, mk, mv, mr, xk, xv, xr);

    // 2) weight conversions
    cvt_kernel<<<dim3((unsigned)(CC / 256)), dim3(256), 0, stream>>>(Wk, wkb);
    cvt_kernel<<<dim3((unsigned)(CC / 256)), dim3(256), 0, stream>>>(Wv, wvb);
    cvt_kernel<<<dim3((unsigned)(CC / 256)), dim3(256), 0, stream>>>(Wr, wrb);
    cvt_kernel<<<dim3((unsigned)(CC / 256)), dim3(256), 0, stream>>>(Wo, wob);

    // 3) k/v/r projections via WMMA (async-LDS double-buffered GEMM)
    dim3 gg(M_ / 128, C_ / 128);   // 64 x 8 workgroups
    gemm_bf16_wmma<<<gg, dim3(256), 0, stream>>>(xk, wkb, kf);
    gemm_bf16_wmma<<<gg, dim3(256), 0, stream>>>(xv, wvb, vf);
    gemm_bf16_wmma<<<gg, dim3(256), 0, stream>>>(xr, wrb, rf);

    // 4) WKV scan + sigmoid gating -> bf16
    wkv_sig_kernel<<<dim3((B_ * C_) / 256), dim3(256), 0, stream>>>(
        td, tf, kf, vf, rf, abf);

    // 5) output projection via WMMA (writes d_out, f32)
    gemm_bf16_wmma<<<gg, dim3(256), 0, stream>>>(abf, wob, out);
}
